// WindowSelfAttention_67095979098861
// MI455X (gfx1250) — matrix operand order repaired and checked by reading, not compiled, not run on hardware
//
#include <hip/hip_runtime.h>
#include <math.h>

typedef __attribute__((ext_vector_type(16))) _Float16 v16h;
typedef __attribute__((ext_vector_type(8)))  float    v8f;

#define D_MODEL 256
#define SEQ     4096
#define BATCH   4
#define WRAD    64           // window radius
#define KSPAN   144          // 16 queries need keys [s0-64, s0+15+64] -> 144
#define PW      160          // KSPAN padded to multiple of 32 for WMMA K-loop
#define SP      (SEQ + 160)  // padded stride for transposed V (4256, mult of 8)

static __device__ __forceinline__ v8f wmma_f16(v16h a, v16h b, v8f c) {
  return __builtin_amdgcn_wmma_f32_16x16x32_f16(false, a, false, b, (short)0, c, false, false);
}

// ---- fragment loaders (CDNA5 wave32 WMMA layouts) --------------------------

// A (16x32, f16) from row-major f32: lane l -> row m=l&15,
// elems 0..15 : K=(l<16?0:16)+e
static __device__ __forceinline__ v16h load_a_f32(const float* __restrict__ src,
                                                  int ld, int row0, int k0) {
  int l = threadIdx.x & 31;
  int kb = (l < 16) ? 0 : 16;
  const float* p = src + (size_t)(row0 + (l & 15)) * ld + k0;
  v16h a;
#pragma unroll
  for (int e = 0; e < 8; ++e) a[e] = (_Float16)p[kb + e];
#pragma unroll
  for (int e = 0; e < 8; ++e) a[8 + e] = (_Float16)p[8 + kb + e];
  return a;
}

// A (16x32, f16) from row-major f16
static __device__ __forceinline__ v16h load_a_h(const _Float16* __restrict__ src,
                                                int ld, int row0, int k0) {
  int l = threadIdx.x & 31;
  int kb = (l < 16) ? 0 : 16;
  const _Float16* p = src + (size_t)(row0 + (l & 15)) * ld + k0;
  v16h a;
#pragma unroll
  for (int e = 0; e < 8; ++e) a[e] = p[kb + e];
#pragma unroll
  for (int e = 0; e < 8; ++e) a[8 + e] = p[8 + kb + e];
  return a;
}

// B (32x16, f16) from k-contiguous ("n-major") f16: B(k,n) = src[n0+n][k0+k].
// Per lane: 16 contiguous halves -> two b128 loads.
static __device__ __forceinline__ v16h load_b_nmajor(const _Float16* __restrict__ src,
                                                     int ld, int k0, int n0) {
  int l = threadIdx.x & 31;
  const _Float16* p = src + (size_t)(n0 + (l & 15)) * ld + k0 + ((l < 16) ? 0 : 16);
  v16h b;
#pragma unroll
  for (int e = 0; e < 16; ++e) b[e] = p[e];
  return b;
}

// B (32x16, f16) = K^T with per-lane row clamp (rows are contiguous in d)
static __device__ __forceinline__ v16h load_bT_h_clamp(const _Float16* __restrict__ src,
                                                       int ld, int k0, int n0,
                                                       int rmin, int rmax) {
  int l = threadIdx.x & 31;
  int r = n0 + (l & 15);
  r = (r < rmin) ? rmin : ((r > rmax) ? rmax : r);
  const _Float16* p = src + (size_t)r * ld + k0 + ((l < 16) ? 0 : 16);
  v16h b;
#pragma unroll
  for (int e = 0; e < 16; ++e) b[e] = p[e];
  return b;
}

// ---- kernel 0a: weight prep — W(f32,[k][n]) -> Wt(f16,[n][k]), 4 matrices --
__global__ __launch_bounds__(256)
void wprep_kernel(const float* __restrict__ W0, const float* __restrict__ W1,
                  const float* __restrict__ W2, const float* __restrict__ W3,
                  _Float16* __restrict__ Wt) {
  const float* W = (blockIdx.y == 0) ? W0 : (blockIdx.y == 1) ? W1
                 : (blockIdx.y == 2) ? W2 : W3;
  _Float16* dst = Wt + (size_t)blockIdx.y * D_MODEL * D_MODEL;
  const int n = blockIdx.x;
  const int k = threadIdx.x;
  dst[(size_t)n * D_MODEL + k] = (_Float16)W[(size_t)k * D_MODEL + n];
}

// ---- kernel 0b: zero the pad columns of transposed V -----------------------
__global__ __launch_bounds__(256)
void vpad_kernel(_Float16* __restrict__ Vt) {
  const int i = blockIdx.x * blockDim.x + threadIdx.x;   // BATCH*D_MODEL*160
  const int p = i % 160;
  const int bd = i / 160;
  const int col = (p < WRAD) ? p : (SEQ + p);            // [0,64) and [4160,4256)
  Vt[(size_t)bd * SP + col] = (_Float16)0;
}

// ---- kernel 1: fused Q/K/V projection --------------------------------------
// Q,K -> row-major f16 ; V -> transposed padded f16 Vt[b][d][64+s]
__global__ __launch_bounds__(256)
void qkv_kernel(const float* __restrict__ x, const _Float16* __restrict__ Wt,
                const float* __restrict__ bq, const float* __restrict__ bk,
                const float* __restrict__ bv,
                _Float16* __restrict__ Qh, _Float16* __restrict__ Kh,
                _Float16* __restrict__ Vt) {
  const int row0 = blockIdx.x * 16;
  const int b    = row0 / SEQ;
  const int sloc = row0 % SEQ;
  const int wave = threadIdx.x >> 5;
  const int l    = threadIdx.x & 31;
  const _Float16* Wqt = Wt;
  const _Float16* Wkt = Wt + 1 * D_MODEL * D_MODEL;
  const _Float16* Wvt = Wt + 2 * D_MODEL * D_MODEL;

  v16h ax[8];
#pragma unroll
  for (int kk = 0; kk < 8; ++kk) ax[kk] = load_a_f32(x, D_MODEL, row0, kk * 32);

#pragma unroll
  for (int t = 0; t < 2; ++t) {
    const int n0 = (wave + t * 8) * 16;
    v8f cq = {}, ck = {}, cv = {};
#pragma unroll
    for (int kk = 0; kk < 8; ++kk) {
      cq = wmma_f16(ax[kk], load_b_nmajor(Wqt, D_MODEL, kk * 32, n0), cq);
      ck = wmma_f16(ax[kk], load_b_nmajor(Wkt, D_MODEL, kk * 32, n0), ck);
      cv = wmma_f16(ax[kk], load_b_nmajor(Wvt, D_MODEL, kk * 32, n0), cv);
    }
    const int n  = n0 + (l & 15);
    const int mb = (l < 16) ? 0 : 8;
    const float vbq = bq[n], vbk = bk[n], vbv = bv[n];
    _Float16* vdst = Vt + ((size_t)b * D_MODEL + n) * SP + WRAD + sloc + mb;
#pragma unroll
    for (int e = 0; e < 8; ++e) {
      size_t o = (size_t)(row0 + mb + e) * D_MODEL + n;
      Qh[o]   = (_Float16)(cq[e] + vbq);
      Kh[o]   = (_Float16)(ck[e] + vbk);
      vdst[e] = (_Float16)(cv[e] + vbv);   // contiguous -> b128 store
    }
  }
}

// ---- kernel 2: windowed attention, one block per 16-query tile -------------
__global__ __launch_bounds__(256)
void attn_kernel(const _Float16* __restrict__ Qh, const _Float16* __restrict__ Kh,
                 const _Float16* __restrict__ Vt, _Float16* __restrict__ Yh) {
  __shared__ float    sS[16 * KSPAN];   // raw scores
  __shared__ _Float16 sP[16 * PW];      // softmax probs, zero-padded K

  const int b    = blockIdx.y;
  const int s0   = blockIdx.x * 16;
  const int key0 = s0 - WRAD;
  const int wave = threadIdx.x >> 5;
  const int l    = threadIdx.x & 31;
  const float scale = 1.0f / 16.0f;     // 1/sqrt(256)

  const _Float16* Qb  = Qh + (size_t)b * SEQ * D_MODEL;
  const _Float16* Kb  = Kh + (size_t)b * SEQ * D_MODEL;
  const _Float16* Vtb = Vt + (size_t)b * D_MODEL * SP;

  // hint the V window into cache while we do QK^T
  __builtin_prefetch(Vtb + (size_t)threadIdx.x * SP + WRAD + key0, 0, 3);

  // Q fragments: loaded once, reused for all score tiles
  v16h aq[8];
#pragma unroll
  for (int kk = 0; kk < 8; ++kk) aq[kk] = load_a_h(Qb, D_MODEL, s0, kk * 32);

  // --- scores: Q(16x256) x K^T over 9 key tiles -----------------------------
  for (int jt = wave; jt < 9; jt += 8) {
    v8f c = {};
#pragma unroll
    for (int kk = 0; kk < 8; ++kk)
      c = wmma_f16(aq[kk], load_bT_h_clamp(Kb, D_MODEL, kk * 32, key0 + jt * 16,
                                           0, SEQ - 1), c);
    const int col = jt * 16 + (l & 15);
    const int mb  = (l < 16) ? 0 : 8;
#pragma unroll
    for (int e = 0; e < 8; ++e) {
      const int m   = mb + e;
      const int key = key0 + col;
      const bool valid = (key >= 0) && (key < SEQ) && (col >= m) && (col <= m + 2 * WRAD);
      sS[m * KSPAN + col] = valid ? c[e] * scale : -1e30f;
    }
  }
  __syncthreads();

  // --- softmax per row ------------------------------------------------------
  if (threadIdx.x < 16) {
    const int m = threadIdx.x;
    float mx = -1e30f;
    for (int j = 0; j < KSPAN; ++j) mx = fmaxf(mx, sS[m * KSPAN + j]);
    float sum = 0.0f;
    for (int j = 0; j < KSPAN; ++j) sum += __expf(sS[m * KSPAN + j] - mx);
    const float inv = 1.0f / sum;
    for (int j = 0; j < KSPAN; ++j)
      sP[m * PW + j] = (_Float16)(__expf(sS[m * KSPAN + j] - mx) * inv);
    for (int j = KSPAN; j < PW; ++j) sP[m * PW + j] = (_Float16)0.0f;
  }
  __syncthreads();

  // --- y = P(16x160) x V_win(160x256), V read from padded transpose ---------
#pragma unroll
  for (int t = 0; t < 2; ++t) {
    const int n0 = (wave + t * 8) * 16;
    v8f c = {};
#pragma unroll
    for (int kk = 0; kk < 5; ++kk) {
      v16h a = load_a_h(sP, PW, 0, kk * 32);
      // B(k,n) = V[key0+k][n] = Vtb[n*SP + WRAD + key0 + k] : contiguous per lane
      v16h bf;
      {
        const _Float16* p = Vtb + (size_t)(n0 + (l & 15)) * SP
                          + WRAD + key0 + kk * 32 + ((l < 16) ? 0 : 16);
#pragma unroll
        for (int e = 0; e < 16; ++e) bf[e] = p[e];
      }
      c = wmma_f16(a, bf, c);
    }
    const int n  = n0 + (l & 15);
    const int mb = (l < 16) ? 0 : 8;
#pragma unroll
    for (int e = 0; e < 8; ++e)
      Yh[((size_t)b * SEQ + s0 + mb + e) * D_MODEL + n] = (_Float16)c[e];
  }
}

// ---- kernel 3: output projection (f16 in -> f32 out) -----------------------
__global__ __launch_bounds__(256)
void oproj_kernel(const _Float16* __restrict__ Yh, const _Float16* __restrict__ Wot,
                  const float* __restrict__ bo, float* __restrict__ out) {
  const int row0 = blockIdx.x * 16;
  const int wave = threadIdx.x >> 5;
  const int l    = threadIdx.x & 31;

  v16h ay[8];
#pragma unroll
  for (int kk = 0; kk < 8; ++kk) ay[kk] = load_a_h(Yh, D_MODEL, row0, kk * 32);

#pragma unroll
  for (int t = 0; t < 2; ++t) {
    const int n0 = (wave + t * 8) * 16;
    v8f c = {};
#pragma unroll
    for (int kk = 0; kk < 8; ++kk)
      c = wmma_f16(ay[kk], load_b_nmajor(Wot, D_MODEL, kk * 32, n0), c);
    const int n  = n0 + (l & 15);
    const int mb = row0 + ((l < 16) ? 0 : 8);
    const float vb = bo[n];
#pragma unroll
    for (int e = 0; e < 8; ++e)
      out[(size_t)(mb + e) * D_MODEL + n] = c[e] + vb;
  }
}

extern "C" void kernel_launch(void* const* d_in, const int* in_sizes, int n_in,
                              void* d_out, int out_size, void* d_ws, size_t ws_size,
                              hipStream_t stream) {
  const float* x  = (const float*)d_in[0];
  const float* Wq = (const float*)d_in[1];
  const float* bq = (const float*)d_in[2];
  const float* Wk = (const float*)d_in[3];
  const float* bk = (const float*)d_in[4];
  const float* Wv = (const float*)d_in[5];
  const float* bv = (const float*)d_in[6];
  const float* Wo = (const float*)d_in[7];
  const float* bo = (const float*)d_in[8];

  const int rows = BATCH * SEQ;                       // 16384
  _Float16* Qh = (_Float16*)d_ws;                     // 8 MB
  _Float16* Kh = Qh + (size_t)rows * D_MODEL;         // 8 MB
  _Float16* Yh = Kh + (size_t)rows * D_MODEL;         // 8 MB
  _Float16* Vt = Yh + (size_t)rows * D_MODEL;         // BATCH*256*SP f16 (~8.7 MB)
  _Float16* Wt = Vt + (size_t)BATCH * D_MODEL * SP;   // 4 * 256*256 f16 (512 KB)

  wprep_kernel<<<dim3(D_MODEL, 4), 256, 0, stream>>>(Wq, Wk, Wv, Wo, Wt);
  vpad_kernel<<<(BATCH * D_MODEL * 160) / 256, 256, 0, stream>>>(Vt);
  qkv_kernel<<<rows / 16, 256, 0, stream>>>(x, Wt, bq, bk, bv, Qh, Kh, Vt);
  attn_kernel<<<dim3(SEQ / 16, BATCH), 256, 0, stream>>>(Qh, Kh, Vt, Yh);
  oproj_kernel<<<rows / 16, 256, 0, stream>>>(Yh, Wt + 3 * D_MODEL * D_MODEL, bo,
                                              (float*)d_out);
}
